// Llama4TextExperts_55035710931365
// MI455X (gfx1250) — matrix operand order repaired and checked by reading, not compiled, not run on hardware
//
#include <hip/hip_runtime.h>

// MoE experts: out[e] = (silu(hs_e @ Wg_e) * (hs_e @ Wu_e)) @ Wd_e
// E=8, H=4096, D=4096, T=1024.  Compute-bound (AI ~460 FLOP/B) -> bf16 WMMA
// with fp32 accumulation (V_WMMA_F32_16X16X32_BF16, wave32).
// Block tile 128(M) x 64(N), K-tile 64 (2 WMMA k-steps per staged tile).
// B staging reads row-pairs so fp32->bf16 packs into v_cvt_pk + ds_store_b32.

#define NUM_EXPERTS 8
#define HIDDEN 4096
#define EXPERT_DIM 4096
#define TOKENS 1024

typedef __attribute__((ext_vector_type(16))) __bf16    v16bf;
typedef __attribute__((ext_vector_type(8)))  float     v8f;
typedef __attribute__((ext_vector_type(8)))  unsigned  v8u;

#define TK  64   // K-tile (two 16x16x32 k-steps)
#define LDS 72   // padded LDS row stride in halves (144B = 36 banks -> conflict-free)

// Native bf16 conversion (lowers to v_cvt_pk_bf16_f32 / v_cvt_bf16 ops).
__device__ __forceinline__ unsigned short f2bf(float f) {
  return __builtin_bit_cast(unsigned short, (__bf16)f);
}
__device__ __forceinline__ unsigned pk2bf(float lo, float hi) {
  return (unsigned)f2bf(lo) | ((unsigned)f2bf(hi) << 16);
}

// Per-ISA fragment K indices (bf16, 16x16x32):
//  A 16x32: VGPR v holds K pair 2v (+8 if v>=4) (+8 for lanes 16-31)
//  B 32x16: VGPR v holds K pair 2v (+8 if v>=4) (+8 for lanes 16-31)
__device__ __forceinline__ int kpA(int v, int hi) {
  return 2 * v + ((v & 4) ? 8 : 0) + (hi ? 8 : 0);
}
__device__ __forceinline__ int kpB(int v, int hi) {
  return 2 * v + ((v & 4) ? 8 : 0) + (hi ? 8 : 0);
}

// ---------------------------------------------------------------------------
// Kernel 1: gate_up GEMM + SiLU fusion; act (bf16) -> d_ws.
// ---------------------------------------------------------------------------
__global__ __launch_bounds__(256) void moe_gateup_silu(
    const float* __restrict__ hs, const float* __restrict__ gup,
    unsigned short* __restrict__ act) {
  __shared__ unsigned short sA[128 * LDS];   // A tile, bf16, [m][k]
  __shared__ unsigned short sBg[64 * LDS];   // gate B tile, bf16, [n][k]
  __shared__ unsigned short sBu[64 * LDS];   // up   B tile, bf16, [n][k]

  const int tid = threadIdx.x;
  const int e   = blockIdx.z;
  const int m0  = blockIdx.y * 128;
  const int n0  = blockIdx.x * 64;

  const int wave = tid >> 5;      // 8 waves (wave32)
  const int lane = tid & 31;
  const int hi   = lane >> 4;
  const int lr   = lane & 15;
  const int wm   = wave & 3;      // 4 x 32 rows = 128
  const int wn   = wave >> 2;     // 2 x 32 cols = 64

  const float* hsE  = hs  + ((size_t)e * TOKENS + m0) * HIDDEN;
  const float* gupE = gup + (size_t)e * HIDDEN * (2 * EXPERT_DIM);

  v8f accG[2][2] = {};
  v8f accU[2][2] = {};

  for (int kk = 0; kk < HIDDEN; kk += TK) {
    // ---- stage A: 128x64 fp32 -> bf16 LDS (paired b32 stores) ------------
#pragma unroll
    for (int i = 0; i < 8; ++i) {
      int idx = tid + i * 256;           // 0..2047
      int r = idx >> 4, c4 = idx & 15;   // row, float4 column
      float4 v = *(const float4*)(hsE + (size_t)r * HIDDEN + kk + c4 * 4);
      unsigned* d = (unsigned*)&sA[r * LDS + c4 * 4];
      d[0] = pk2bf(v.x, v.y);
      d[1] = pk2bf(v.z, v.w);
    }
    // ---- stage B (gate & up): 64x64 fp32 -> bf16 LDS transposed [n][k] ---
    // Row-pair loads: K-pair (2k,2k+1) at a column packs into one b32 store.
#pragma unroll
    for (int i = 0; i < 2; ++i) {
      int idx = tid + i * 256;           // 0..511
      int krp = idx >> 4, c4 = idx & 15; // k row-pair (0..31), float4 column
      const float* gp = gupE + (size_t)(kk + 2 * krp) * (2 * EXPERT_DIM) + n0 + c4 * 4;
      float4 g0 = *(const float4*)(gp);
      float4 g1 = *(const float4*)(gp + 2 * EXPERT_DIM);
      float4 u0 = *(const float4*)(gp + EXPERT_DIM);
      float4 u1 = *(const float4*)(gp + EXPERT_DIM + 2 * EXPERT_DIM);
      int nb = c4 * 4, kc = 2 * krp;
      *(unsigned*)&sBg[(nb + 0) * LDS + kc] = pk2bf(g0.x, g1.x);
      *(unsigned*)&sBg[(nb + 1) * LDS + kc] = pk2bf(g0.y, g1.y);
      *(unsigned*)&sBg[(nb + 2) * LDS + kc] = pk2bf(g0.z, g1.z);
      *(unsigned*)&sBg[(nb + 3) * LDS + kc] = pk2bf(g0.w, g1.w);
      *(unsigned*)&sBu[(nb + 0) * LDS + kc] = pk2bf(u0.x, u1.x);
      *(unsigned*)&sBu[(nb + 1) * LDS + kc] = pk2bf(u0.y, u1.y);
      *(unsigned*)&sBu[(nb + 2) * LDS + kc] = pk2bf(u0.z, u1.z);
      *(unsigned*)&sBu[(nb + 3) * LDS + kc] = pk2bf(u0.w, u1.w);
    }
    if (kk + TK < HIDDEN)   // gfx1250 global_prefetch_b8 of next A tile
      __builtin_prefetch(hsE + (size_t)(tid >> 1) * HIDDEN + kk + TK, 0, 1);
    __syncthreads();

    // ---- two k-steps: 8 WMMAs each; fragments dead between steps ---------
#pragma unroll
    for (int t = 0; t < 2; ++t) {
      int kb = t * 32;
      v8u aU[2], gU[2], uU[2];
#pragma unroll
      for (int s = 0; s < 2; ++s) {
        int row = wm * 32 + s * 16 + lr;
        int col = wn * 32 + s * 16 + lr;
#pragma unroll
        for (int v = 0; v < 8; ++v) {
          aU[s][v] = *(const unsigned*)&sA[row * LDS + kb + kpA(v, hi)];
          gU[s][v] = *(const unsigned*)&sBg[col * LDS + kb + kpB(v, hi)];
          uU[s][v] = *(const unsigned*)&sBu[col * LDS + kb + kpB(v, hi)];
        }
      }
      v16bf aF[2], bG[2], bU[2];
#pragma unroll
      for (int s = 0; s < 2; ++s) {
        aF[s] = __builtin_bit_cast(v16bf, aU[s]);
        bG[s] = __builtin_bit_cast(v16bf, gU[s]);
        bU[s] = __builtin_bit_cast(v16bf, uU[s]);
      }
#pragma unroll
      for (int sm = 0; sm < 2; ++sm)
#pragma unroll
        for (int sn = 0; sn < 2; ++sn) {
          accG[sm][sn] = __builtin_amdgcn_wmma_f32_16x16x32_bf16(
              false, aF[sm], false, bG[sn], (short)0, accG[sm][sn], false, false);
          accU[sm][sn] = __builtin_amdgcn_wmma_f32_16x16x32_bf16(
              false, aF[sm], false, bU[sn], (short)0, accU[sm][sn], false, false);
        }
    }
    __syncthreads();
  }

  // ---- epilogue: act = up * silu(gate), bf16 to workspace ----------------
  unsigned short* actE = act + ((size_t)e * TOKENS + m0) * EXPERT_DIM;
#pragma unroll
  for (int sm = 0; sm < 2; ++sm)
#pragma unroll
    for (int sn = 0; sn < 2; ++sn)
#pragma unroll
      for (int r = 0; r < 8; ++r) {
        float g = accG[sm][sn][r];
        float u = accU[sm][sn][r];
        float a = u * (g / (1.0f + __expf(-g)));
        int row = wm * 32 + sm * 16 + r + hi * 8;   // D: M=r(+8), N=lane&15
        int col = n0 + wn * 32 + sn * 16 + lr;
        actE[(size_t)row * EXPERT_DIM + col] = f2bf(a);
      }
}

// ---------------------------------------------------------------------------
// Kernel 2: out = act(bf16) @ down_proj.  Block tile 128(M) x 64(N), K-tile 64.
// ---------------------------------------------------------------------------
__global__ __launch_bounds__(256) void moe_down(
    const unsigned short* __restrict__ act, const float* __restrict__ dp,
    float* __restrict__ out) {
  __shared__ unsigned short sA[128 * LDS];
  __shared__ unsigned short sB[64 * LDS];

  const int tid = threadIdx.x;
  const int e   = blockIdx.z;
  const int m0  = blockIdx.y * 128;
  const int n0  = blockIdx.x * 64;

  const int wave = tid >> 5;
  const int lane = tid & 31;
  const int hi   = lane >> 4;
  const int lr   = lane & 15;
  const int wm   = wave & 3;
  const int wn   = wave >> 2;

  const unsigned short* actE = act + ((size_t)e * TOKENS + m0) * EXPERT_DIM;
  const float* dpE = dp + (size_t)e * EXPERT_DIM * HIDDEN;

  v8f acc[2][2] = {};

  for (int kk = 0; kk < EXPERT_DIM; kk += TK) {
    // A: 128x64 bf16, straight copy to LDS (8B packets)
#pragma unroll
    for (int i = 0; i < 8; ++i) {
      int idx = tid + i * 256;           // 0..2047
      int r = idx >> 4, c = idx & 15;    // row, 4-half column
      uint2 v = *(const uint2*)(actE + (size_t)r * EXPERT_DIM + kk + c * 4);
      *(uint2*)&sA[r * LDS + c * 4] = v;
    }
    // B: 64x64 fp32 -> bf16, transposed [n][k], row-pair packed stores
#pragma unroll
    for (int i = 0; i < 2; ++i) {
      int idx = tid + i * 256;           // 0..511
      int krp = idx >> 4, c4 = idx & 15;
      const float* bp = dpE + (size_t)(kk + 2 * krp) * HIDDEN + n0 + c4 * 4;
      float4 b0 = *(const float4*)(bp);
      float4 b1 = *(const float4*)(bp + HIDDEN);
      int nb = c4 * 4, kc = 2 * krp;
      *(unsigned*)&sB[(nb + 0) * LDS + kc] = pk2bf(b0.x, b1.x);
      *(unsigned*)&sB[(nb + 1) * LDS + kc] = pk2bf(b0.y, b1.y);
      *(unsigned*)&sB[(nb + 2) * LDS + kc] = pk2bf(b0.z, b1.z);
      *(unsigned*)&sB[(nb + 3) * LDS + kc] = pk2bf(b0.w, b1.w);
    }
    if (kk + TK < EXPERT_DIM)
      __builtin_prefetch(dpE + (size_t)(kk + TK + (tid >> 3)) * HIDDEN + n0, 0, 1);
    __syncthreads();

#pragma unroll
    for (int t = 0; t < 2; ++t) {
      int kb = t * 32;
      v8u aU[2], bUv[2];
#pragma unroll
      for (int s = 0; s < 2; ++s) {
        int row = wm * 32 + s * 16 + lr;
        int col = wn * 32 + s * 16 + lr;
#pragma unroll
        for (int v = 0; v < 8; ++v) {
          aU[s][v]  = *(const unsigned*)&sA[row * LDS + kb + kpA(v, hi)];
          bUv[s][v] = *(const unsigned*)&sB[col * LDS + kb + kpB(v, hi)];
        }
      }
      v16bf aF[2], bF[2];
#pragma unroll
      for (int s = 0; s < 2; ++s) {
        aF[s] = __builtin_bit_cast(v16bf, aU[s]);
        bF[s] = __builtin_bit_cast(v16bf, bUv[s]);
      }
#pragma unroll
      for (int sm = 0; sm < 2; ++sm)
#pragma unroll
        for (int sn = 0; sn < 2; ++sn)
          acc[sm][sn] = __builtin_amdgcn_wmma_f32_16x16x32_bf16(
              false, aF[sm], false, bF[sn], (short)0, acc[sm][sn], false, false);
    }
    __syncthreads();
  }

  float* outE = out + ((size_t)e * TOKENS + m0) * HIDDEN;
#pragma unroll
  for (int sm = 0; sm < 2; ++sm)
#pragma unroll
    for (int sn = 0; sn < 2; ++sn)
#pragma unroll
      for (int r = 0; r < 8; ++r) {
        int row = wm * 32 + sm * 16 + r + hi * 8;
        int col = n0 + wn * 32 + sn * 16 + lr;
        outE[(size_t)row * HIDDEN + col] = acc[sm][sn][r];
      }
}

extern "C" void kernel_launch(void* const* d_in, const int* in_sizes, int n_in,
                              void* d_out, int out_size, void* d_ws, size_t ws_size,
                              hipStream_t stream) {
  (void)in_sizes; (void)n_in; (void)out_size; (void)ws_size;
  const float* hs  = (const float*)d_in[0];
  const float* gup = (const float*)d_in[1];
  const float* dp  = (const float*)d_in[2];
  float* out = (float*)d_out;
  unsigned short* actws = (unsigned short*)d_ws;  // 8*1024*4096 bf16 = 64 MiB

  dim3 grid1(EXPERT_DIM / 64, TOKENS / 128, NUM_EXPERTS);
  dim3 grid2(HIDDEN / 64, TOKENS / 128, NUM_EXPERTS);
  moe_gateup_silu<<<grid1, 256, 0, stream>>>(hs, gup, actws);
  moe_down<<<grid2, 256, 0, stream>>>(actws, dp, out);
}
